// CustomConv_24206435680442
// MI455X (gfx1250) — compile-verified, matrix operand order checked by reading, not run on hardware
//
#include <hip/hip_runtime.h>

typedef __attribute__((ext_vector_type(16))) __bf16          v16bf;
typedef __attribute__((ext_vector_type(8)))  __bf16          v8bf;
typedef __attribute__((ext_vector_type(4)))  __bf16          v4bf;
typedef __attribute__((ext_vector_type(8)))  float           v8f;
typedef __attribute__((ext_vector_type(4)))  float           v4f;

#define IMG      56
#define PIMG     58                     // padded image side
#define CIN      128
#define COUT     256
#define HWPIX    (IMG * IMG)            // 3136
#define KTOT     1152
#define KCHUNKS  36                     // 1152 / 32
#define LDSPITCH 40                     // 32 k-elems + 8 pad (80B rows, 16B aligned)

// ---------------- common helpers ----------------

// Build a 16x(K=32) bf16 WMMA fragment from LDS: two 16B chunks, 16 elems apart.
static __device__ __forceinline__ v16bf frag16(const __bf16* p) {
    union { v16bf full; v8bf h[2]; } f;
    f.h[0] = *(const v8bf*)(p);
    f.h[1] = *(const v8bf*)(p + 16);
    return f.full;
}

// CDNA5 async copy: 16B global -> LDS, tracked by ASYNCcnt (bypasses VGPRs).
static __device__ __forceinline__ void async_ld16(unsigned lds, const void* g) {
    asm volatile("global_load_async_to_lds_b128 %0, %1, off"
                 :: "v"(lds), "v"(g) : "memory");
}
static __device__ __forceinline__ void wait_async0() {
    asm volatile("s_wait_asynccnt 0x0" ::: "memory");
}
static __device__ __forceinline__ unsigned lds_off(const void* p) {
    return (unsigned)(unsigned long long)p;   // LDS generic addr: offset in low 32b
}

// ---------------- pre-pass: convert + pad input, convert + transpose weights ----

__global__ void cvt_pad_x(const float* __restrict__ x,
                          unsigned short* __restrict__ xHiU,
                          unsigned short* __restrict__ xLoU) {
    __bf16* xHi = (__bf16*)xHiU;
    __bf16* xLo = (__bf16*)xLoU;
    const int idx = blockIdx.x * 256 + threadIdx.x;     // 32*58*58*16 threads
    const int c8  = (idx & 15) * 8;
    const int r   = idx >> 4;                           // n*58*58 + ph*58 + pw
    const int pw  = r % PIMG;
    const int r2  = r / PIMG;
    const int ph  = r2 % PIMG;
    const int n   = r2 / PIMG;

    float v[8];
    const bool inside = (ph >= 1) && (ph <= IMG) && (pw >= 1) && (pw <= IMG);
    if (inside) {
        const float* s = x + (((size_t)(n * IMG + (ph - 1)) * IMG + (pw - 1)) * CIN + c8);
        v4f a = *(const v4f*)(s);
        v4f b = *(const v4f*)(s + 4);
#pragma unroll
        for (int j = 0; j < 4; ++j) { v[j] = a[j]; v[4 + j] = b[j]; }
    } else {
#pragma unroll
        for (int j = 0; j < 8; ++j) v[j] = 0.0f;
    }
    v8bf hi, lo;
#pragma unroll
    for (int j = 0; j < 8; ++j) {
        __bf16 h = (__bf16)v[j];
        hi[j] = h;
        lo[j] = (__bf16)(v[j] - (float)h);
    }
    const size_t o = (size_t)r * CIN + c8;
    *(v8bf*)(xHi + o) = hi;
    *(v8bf*)(xLo + o) = lo;
}

__global__ void cvt_w(const float* __restrict__ w,
                      unsigned short* __restrict__ wHiU,
                      unsigned short* __restrict__ wLoU) {
    __bf16* wHi = (__bf16*)wHiU;
    __bf16* wLo = (__bf16*)wLoU;
    const int idx = blockIdx.x * 256 + threadIdx.x;     // 1152*64 threads
    const int k   = idx >> 6;
    const int co4 = (idx & 63) * 4;
    v4f r = *(const v4f*)(w + (size_t)k * COUT + co4);
#pragma unroll
    for (int c = 0; c < 4; ++c) {
        __bf16 h = (__bf16)r[c];
        wHi[(size_t)(co4 + c) * KTOT + k] = h;
        wLo[(size_t)(co4 + c) * KTOT + k] = (__bf16)(r[c] - (float)h);
    }
}

// ---------------- main kernel: async-staged implicit GEMM, bf16x3 ----------------

__launch_bounds__(256, 2)
__global__ void conv3x3_wmma_async(const unsigned short* __restrict__ xHiU,
                                   const unsigned short* __restrict__ xLoU,
                                   const unsigned short* __restrict__ wHiU,
                                   const unsigned short* __restrict__ wLoU,
                                   const float* __restrict__ bias,
                                   float* __restrict__ out) {
    // [buf][plane(hi/lo)][row*40] ; 80 KB total
    __shared__ __bf16 sA[2][2][128 * LDSPITCH];
    __shared__ __bf16 sB[2][2][128 * LDSPITCH];

    const int t = threadIdx.x;

    // ---------- staging roles (plane is wave-uniform) ----------
    const int plane = t >> 7;                       // 0=hi, 1=lo
    const int aRow  = t & 127;
    const int m     = blockIdx.x * 128 + aRow;
    const int nimg  = m / HWPIX;
    const int rem   = m % HWPIX;
    const int oh    = rem / IMG;
    const int ow    = rem % IMG;
    const __bf16* xP = (const __bf16*)(plane ? xLoU : xHiU);
    const int bCo   = t & 127;
    const __bf16* wPB = (const __bf16*)(plane ? wLoU : wHiU)
                        + (size_t)(blockIdx.y * 128 + bCo) * KTOT;

    unsigned aLds[2], bLds[2];
#pragma unroll
    for (int b = 0; b < 2; ++b) {
        aLds[b] = lds_off(&sA[b][plane][aRow * LDSPITCH]);
        bLds[b] = lds_off(&sB[b][plane][bCo  * LDSPITCH]);
    }

    auto stage = [&](int kc, int buf) {
        const int tap = kc >> 2;                    // filter tap 0..8
        const int c0  = (kc & 3) * 32;
        const int fy  = tap / 3, fx = tap % 3;
        // padded image: always in-bounds
        const __bf16* ga = xP + (((size_t)(nimg * PIMG + oh + fy)) * PIMG + (ow + fx)) * CIN + c0;
        const __bf16* gb = wPB + kc * 32;
#pragma unroll
        for (int i = 0; i < 4; ++i)
            async_ld16(aLds[buf] + i * 16, ga + i * 8);
#pragma unroll
        for (int i = 0; i < 4; ++i)
            async_ld16(bLds[buf] + i * 16, gb + i * 8);
    };

    // ---------- compute roles ----------
    const int lane = t & 31;
    const int wave = t >> 5;
    const int wmi  = wave & 1;       // wave M offset: *64
    const int wni  = wave >> 1;      // wave N offset: *32  (0..3)
    const int lrow = lane & 15;
    const int e0   = ((lane >> 4) & 1) * 8;

    v8f acc[4][2];
#pragma unroll
    for (int mt = 0; mt < 4; ++mt)
#pragma unroll
        for (int nt = 0; nt < 2; ++nt) acc[mt][nt] = (v8f)(0.0f);

    stage(0, 0);
    wait_async0();
    __syncthreads();

#pragma unroll 1
    for (int kc = 0; kc < KCHUNKS; ++kc) {
        const int cur = kc & 1;
        if (kc + 1 < KCHUNKS)
            stage(kc + 1, cur ^ 1);   // async fill of the other buffer

        v16bf Bh[2], Bl[2];
#pragma unroll
        for (int nt = 0; nt < 2; ++nt) {
            const int bb = (wni * 32 + nt * 16 + lrow) * LDSPITCH + e0;
            Bh[nt] = frag16(&sB[cur][0][bb]);
            Bl[nt] = frag16(&sB[cur][1][bb]);
        }
#pragma unroll
        for (int mt = 0; mt < 4; ++mt) {
            const int ab = (wmi * 64 + mt * 16 + lrow) * LDSPITCH + e0;
            v16bf Ah = frag16(&sA[cur][0][ab]);
            v16bf Al = frag16(&sA[cur][1][ab]);
#pragma unroll
            for (int nt = 0; nt < 2; ++nt) {
                acc[mt][nt] = __builtin_amdgcn_wmma_f32_16x16x32_bf16(
                    false, Al, false, Bh[nt], (short)0, acc[mt][nt], false, false);
                acc[mt][nt] = __builtin_amdgcn_wmma_f32_16x16x32_bf16(
                    false, Ah, false, Bl[nt], (short)0, acc[mt][nt], false, false);
                acc[mt][nt] = __builtin_amdgcn_wmma_f32_16x16x32_bf16(
                    false, Ah, false, Bh[nt], (short)0, acc[mt][nt], false, false);
            }
        }

        if (kc + 1 < KCHUNKS)
            wait_async0();            // next buffer fully in LDS
        __syncthreads();
    }

    // ---------- epilogue: bias + ReLU + store ----------
#pragma unroll
    for (int nt = 0; nt < 2; ++nt) {
        const int cc = blockIdx.y * 128 + wni * 32 + nt * 16 + lrow;
        const float bv = bias[cc];
#pragma unroll
        for (int mt = 0; mt < 4; ++mt) {
            const int mrow = blockIdx.x * 128 + wmi * 64 + mt * 16 + ((lane >> 4) << 3);
            float* o = out + (size_t)mrow * COUT + cc;
#pragma unroll
            for (int r = 0; r < 8; ++r) {
                float v = acc[mt][nt][r] + bv;
                o[(size_t)r * COUT] = fmaxf(v, 0.0f);
            }
        }
    }
}

// ---------------- fallback: self-converting kernel (small d_ws) ----------------

__launch_bounds__(256, 2)
__global__ void conv3x3_wmma_fallback(const float* __restrict__ x,
                                      const float* __restrict__ wgt,
                                      const float* __restrict__ bias,
                                      float* __restrict__ out) {
    __shared__ __bf16 sAhi[2][128 * LDSPITCH];
    __shared__ __bf16 sAlo[2][128 * LDSPITCH];
    __shared__ __bf16 sBhi[2][128 * LDSPITCH];
    __shared__ __bf16 sBlo[2][128 * LDSPITCH];

    const int t = threadIdx.x;
    const int aRow  = t & 127;
    const int aHalf = t >> 7;
    const int m     = blockIdx.x * 128 + aRow;
    const int nimg  = m / HWPIX;
    const int rem   = m % HWPIX;
    const int oh    = rem / IMG;
    const int ow    = rem % IMG;
    const int bCo4  = (t & 31) * 4;
    const int bKg   = t >> 5;
    const float* wB = wgt + (size_t)(blockIdx.y * 128 + bCo4);

    float aReg[16], bReg[16];

    auto loadA = [&](int kc) {
        const int tap = kc >> 2;
        const int c0  = (kc & 3) * 32;
        const int fy  = tap / 3, fx = tap % 3;
        const int ih  = oh + fy - 1, iw = ow + fx - 1;
        const bool valid = ((unsigned)ih < (unsigned)IMG) && ((unsigned)iw < (unsigned)IMG);
        if (valid) {
            const float* src = x + (((size_t)(nimg * IMG + ih) * IMG + iw) * CIN
                                    + c0 + aHalf * 16);
            v4f v0 = *(const v4f*)(src + 0);
            v4f v1 = *(const v4f*)(src + 4);
            v4f v2 = *(const v4f*)(src + 8);
            v4f v3 = *(const v4f*)(src + 12);
#pragma unroll
            for (int j = 0; j < 4; ++j) {
                aReg[j] = v0[j]; aReg[4 + j] = v1[j];
                aReg[8 + j] = v2[j]; aReg[12 + j] = v3[j];
            }
        } else {
#pragma unroll
            for (int j = 0; j < 16; ++j) aReg[j] = 0.0f;
        }
    };
    auto loadB = [&](int kc) {
        const float* src = wB + (size_t)(kc * 32 + bKg * 4) * COUT;
#pragma unroll
        for (int j = 0; j < 4; ++j) {
            v4f r = *(const v4f*)(src + (size_t)j * COUT);
#pragma unroll
            for (int c = 0; c < 4; ++c) bReg[j * 4 + c] = r[c];
        }
    };
    auto cvtStore = [&](int buf) {
        v8bf h0, h1, l0, l1;
#pragma unroll
        for (int j = 0; j < 8; ++j) {
            __bf16 hiA = (__bf16)aReg[j];
            h0[j] = hiA; l0[j] = (__bf16)(aReg[j] - (float)hiA);
            __bf16 hiB = (__bf16)aReg[8 + j];
            h1[j] = hiB; l1[j] = (__bf16)(aReg[8 + j] - (float)hiB);
        }
        const int ab = aRow * LDSPITCH + aHalf * 16;
        *(v8bf*)&sAhi[buf][ab] = h0;  *(v8bf*)&sAhi[buf][ab + 8] = h1;
        *(v8bf*)&sAlo[buf][ab] = l0;  *(v8bf*)&sAlo[buf][ab + 8] = l1;
#pragma unroll
        for (int c = 0; c < 4; ++c) {
            v4bf hb, lb;
#pragma unroll
            for (int j = 0; j < 4; ++j) {
                float v = bReg[j * 4 + c];
                __bf16 hi = (__bf16)v;
                hb[j] = hi; lb[j] = (__bf16)(v - (float)hi);
            }
            const int bb = (bCo4 + c) * LDSPITCH + bKg * 4;
            *(v4bf*)&sBhi[buf][bb] = hb;
            *(v4bf*)&sBlo[buf][bb] = lb;
        }
    };

    const int lane = t & 31;
    const int wave = t >> 5;
    const int wmi  = wave & 1;
    const int wni  = wave >> 1;
    const int lrow = lane & 15;
    const int e0   = ((lane >> 4) & 1) * 8;

    v8f acc[4][2];
#pragma unroll
    for (int mt = 0; mt < 4; ++mt)
#pragma unroll
        for (int nt = 0; nt < 2; ++nt) acc[mt][nt] = (v8f)(0.0f);

    loadA(0); loadB(0); cvtStore(0);

#pragma unroll 1
    for (int kc = 0; kc < KCHUNKS; ++kc) {
        const int cur = kc & 1;
        if (kc + 1 < KCHUNKS) { loadA(kc + 1); loadB(kc + 1); }
        __syncthreads();

        v16bf Bh[2], Bl[2];
#pragma unroll
        for (int nt = 0; nt < 2; ++nt) {
            const int bb = (wni * 32 + nt * 16 + lrow) * LDSPITCH + e0;
            Bh[nt] = frag16(&sBhi[cur][bb]);
            Bl[nt] = frag16(&sBlo[cur][bb]);
        }
#pragma unroll
        for (int mt = 0; mt < 4; ++mt) {
            const int ab = (wmi * 64 + mt * 16 + lrow) * LDSPITCH + e0;
            v16bf Ah = frag16(&sAhi[cur][ab]);
            v16bf Al = frag16(&sAlo[cur][ab]);
#pragma unroll
            for (int nt = 0; nt < 2; ++nt) {
                acc[mt][nt] = __builtin_amdgcn_wmma_f32_16x16x32_bf16(
                    false, Al, false, Bh[nt], (short)0, acc[mt][nt], false, false);
                acc[mt][nt] = __builtin_amdgcn_wmma_f32_16x16x32_bf16(
                    false, Ah, false, Bl[nt], (short)0, acc[mt][nt], false, false);
                acc[mt][nt] = __builtin_amdgcn_wmma_f32_16x16x32_bf16(
                    false, Ah, false, Bh[nt], (short)0, acc[mt][nt], false, false);
            }
        }
        if (kc + 1 < KCHUNKS) cvtStore(cur ^ 1);
    }

#pragma unroll
    for (int nt = 0; nt < 2; ++nt) {
        const int cc = blockIdx.y * 128 + wni * 32 + nt * 16 + lrow;
        const float bv = bias[cc];
#pragma unroll
        for (int mt = 0; mt < 4; ++mt) {
            const int mrow = blockIdx.x * 128 + wmi * 64 + mt * 16 + ((lane >> 4) << 3);
            float* o = out + (size_t)mrow * COUT + cc;
#pragma unroll
            for (int r = 0; r < 8; ++r) {
                float v = acc[mt][nt][r] + bv;
                o[(size_t)r * COUT] = fmaxf(v, 0.0f);
            }
        }
    }
}

// ---------------- launch ----------------

extern "C" void kernel_launch(void* const* d_in, const int* in_sizes, int n_in,
                              void* d_out, int out_size, void* d_ws, size_t ws_size,
                              hipStream_t stream) {
    (void)in_sizes; (void)n_in; (void)out_size;
    const float* x    = (const float*)d_in[0];   // (32,56,56,128)
    const float* wgt  = (const float*)d_in[1];   // (3,3,128,256)
    const float* bias = (const float*)d_in[2];   // (1,1,1,256)
    float* out        = (float*)d_out;           // (32,56,56,256)

    const size_t SX = (size_t)32 * PIMG * PIMG * CIN;   // padded plane elements
    const size_t SW = (size_t)KTOT * COUT;
    const size_t need = (2 * SX + 2 * SW) * 2;          // bytes (bf16)

    dim3 grid(784, 2, 1);
    dim3 block(256, 1, 1);

    if (d_ws && ws_size >= need) {
        unsigned short* xHi = (unsigned short*)d_ws;
        unsigned short* xLo = xHi + SX;
        unsigned short* wHi = xLo + SX;
        unsigned short* wLo = wHi + SW;
        hipLaunchKernelGGL(cvt_pad_x, dim3((32 * PIMG * PIMG * 16) / 256), block, 0, stream,
                           x, xHi, xLo);
        hipLaunchKernelGGL(cvt_w, dim3((KTOT * 64) / 256), block, 0, stream,
                           wgt, wHi, wLo);
        hipLaunchKernelGGL(conv3x3_wmma_async, grid, block, 0, stream,
                           xHi, xLo, wHi, wLo, bias, out);
    } else {
        hipLaunchKernelGGL(conv3x3_wmma_fallback, grid, block, 0, stream,
                           x, wgt, bias, out);
    }
}
